// VQVAE_15152644620358
// MI455X (gfx1250) — matrix operand order repaired and checked by reading, not compiled, not run on hardware
//
#include <hip/hip_runtime.h>
#include <hip/hip_bf16.h>

typedef __attribute__((ext_vector_type(16))) __bf16 v16bf;
typedef __attribute__((ext_vector_type(8)))  float  v8f;
typedef __attribute__((ext_vector_type(4)))  float  v4f;

#define BATCH      65536
#define IN_DIM     1024
#define LAT        100
#define NE         21
#define NT         7      /* N tiles of 16 -> 112 >= 100 */
#define BS_PITCH   34     /* ushort pitch for Bs rows (pad vs bank conflicts) */
#define ZP         105    /* float pitch for z_e rows (odd -> conflict free) */
#define WAVES      4      /* 4 waves x 32 rows = 128 rows per WG */

__device__ __forceinline__ unsigned short f2bf(float f) {
    unsigned int u = __float_as_uint(f);
    unsigned int r = u + 0x7FFFu + ((u >> 16) & 1u);   // round-to-nearest-even
    return (unsigned short)(r >> 16);
}

union AFu { v16bf v; unsigned short s[16]; };

__device__ __forceinline__ void make_afrag(AFu& af, const float* p) {
    // p points at xrow + kt + aklo; elements are 16 contiguous k (k = aklo + s)
    const v4f* p0 = (const v4f*)p;
    const v4f* p1 = (const v4f*)(p + 8);
    v4f f0 = p0[0], f1 = p0[1], f2 = p1[0], f3 = p1[1];
    af.s[0]  = f2bf(f0.x); af.s[1]  = f2bf(f0.y);
    af.s[2]  = f2bf(f0.z); af.s[3]  = f2bf(f0.w);
    af.s[4]  = f2bf(f1.x); af.s[5]  = f2bf(f1.y);
    af.s[6]  = f2bf(f1.z); af.s[7]  = f2bf(f1.w);
    af.s[8]  = f2bf(f2.x); af.s[9]  = f2bf(f2.y);
    af.s[10] = f2bf(f2.z); af.s[11] = f2bf(f2.w);
    af.s[12] = f2bf(f3.x); af.s[13] = f2bf(f3.y);
    af.s[14] = f2bf(f3.z); af.s[15] = f2bf(f3.w);
}

// ---------------------------------------------------------------------------
// table[e][j] = sum_k codebook[e,k] * W_dec[j,k] + b_dec[j]   (21 x 1024)
// ---------------------------------------------------------------------------
__global__ __launch_bounds__(256) void recon_table_kernel(
    const float* __restrict__ codebook, const float* __restrict__ W_dec,
    const float* __restrict__ b_dec, float* __restrict__ table)
{
    const int e = blockIdx.x >> 2;                       // 21 codes
    const int j = ((blockIdx.x & 3) << 8) + threadIdx.x; // 0..1023
    const float* cb = codebook + e * LAT;
    const float* w  = W_dec + (size_t)j * LAT;
    float acc = b_dec[j];
#pragma unroll 4
    for (int k = 0; k < LAT; ++k) acc = fmaf(cb[k], w[k], acc);
    table[e * IN_DIM + j] = acc;
}

// ---------------------------------------------------------------------------
// Fused encoder GEMM (bf16 WMMA) + VQ argmin + loss partials + index write.
// 128 threads = 4 waves; each wave computes a 32x112 slab (2 M-tiles reuse
// each B fragment -> half the LDS-read bytes per FLOP); WG covers 128 rows.
// ---------------------------------------------------------------------------
__global__ __launch_bounds__(128, 1) void encoder_vq_kernel(
    const float* __restrict__ x, const float* __restrict__ W_enc,
    const float* __restrict__ b_enc, const float* __restrict__ codebook,
    int* __restrict__ idx_out, float* __restrict__ partial)
{
    __shared__ unsigned short Bs[112 * BS_PITCH];   // W_enc k-tile, bf16
    __shared__ float zs[WAVES][32][ZP];             // z_e per wave (32 rows)
    __shared__ float wsum[WAVES];

    const int tid  = threadIdx.x;
    const int lane = tid & 31;
    const int wave = tid >> 5;
    const int rowBase = blockIdx.x * 128;

    // A-fragment lane mapping (16-bit A 16x32, row = lane%16, contiguous k = 16*(lane/16) + s)
    const int am   = lane & 15;
    const int aklo = (lane >> 4) << 4;   // 0 or 16
    const float* xrow0 = x + (size_t)(rowBase + wave * 32 + am) * IN_DIM;
    const float* xrow1 = xrow0 + 16 * (size_t)IN_DIM;

    v8f c[2][NT] = {};

    for (int kt = 0; kt < IN_DIM; kt += 32) {
        __syncthreads();   // all waves done reading previous Bs
        // ---- cooperative stage of W_enc[:,kt..kt+31] as bf16 (112x32) ----
#pragma unroll
        for (int i = 0; i < 28; ++i) {
            int e = i * 128 + tid;          // 0..3583
            int n = e >> 5, k = e & 31;
            float v = (n < LAT) ? W_enc[n * IN_DIM + kt + k] : 0.0f;
            Bs[n * BS_PITCH + k] = f2bf(v);
        }
        __syncthreads();

        if (kt + 32 < IN_DIM) {
            __builtin_prefetch(xrow0 + kt + 32, 0, 0);
            __builtin_prefetch(xrow1 + kt + 32, 0, 0);
        }

        // ---- two A fragments (rows m..m+15 and m+16..m+31 of the slab) ----
        AFu af0, af1;
        make_afrag(af0, xrow0 + kt + aklo);
        make_afrag(af1, xrow1 + kt + aklo);

        // ---- B fragments from LDS; each feeds 2 WMMAs ----
        const int bn = lane & 15;
        const int bk = (lane >> 4) << 4;   // 0 or 16
#pragma unroll
        for (int t = 0; t < NT; ++t) {
            union { v16bf v; unsigned int u[8]; } bf;
            const unsigned int* br =
                (const unsigned int*)&Bs[(t * 16 + bn) * BS_PITCH + bk];
#pragma unroll
            for (int j = 0; j < 8; ++j) bf.u[j] = br[j];
            c[0][t] = __builtin_amdgcn_wmma_f32_16x16x32_bf16(
                false, af0.v, false, bf.v, (short)0, c[0][t], false, false);
            c[1][t] = __builtin_amdgcn_wmma_f32_16x16x32_bf16(
                false, af1.v, false, bf.v, (short)0, c[1][t], false, false);
        }
    }

    // ---- epilogue: add bias, spill z_e slab to LDS (C layout: ISA 7.12.2) --
    const int mhalf = (lane >> 4) << 3;
#pragma unroll
    for (int mt = 0; mt < 2; ++mt) {
#pragma unroll
        for (int t = 0; t < NT; ++t) {
            int n = t * 16 + (lane & 15);
            if (n < LAT) {
                float bias = b_enc[n];
#pragma unroll
                for (int v = 0; v < 8; ++v)
                    zs[wave][mt * 16 + mhalf + v][n] = c[mt][t][v] + bias;
            }
        }
    }
    // LDS ops are in-order within a wave; zs is wave-private. No barrier needed.

    // ---- per-row argmin over 21 codes (one row per lane, all 32 active);
    //      min distance == per-row loss sum -------------------------------
    const float* zrow = zs[wave][lane];
    float zn = 0.0f;
#pragma unroll 4
    for (int n = 0; n < LAT; ++n) { float z = zrow[n]; zn = fmaf(z, z, zn); }
    float best = 3.402823e38f; int bi = 0;
    for (int e = 0; e < NE; ++e) {
        const float* cb = codebook + e * LAT;
        float dot = 0.0f, cn = 0.0f;
#pragma unroll 4
        for (int n = 0; n < LAT; ++n) {
            float cv = cb[n];
            dot = fmaf(zrow[n], cv, dot);
            cn  = fmaf(cv, cv, cn);
        }
        float dist = zn + cn - 2.0f * dot;
        if (dist < best) { best = dist; bi = e; }
    }
    idx_out[rowBase + wave * 32 + lane] = bi;

    // wave32 reduction of loss contributions (deterministic within wave)
    float contrib = best;
    for (int off = 16; off > 0; off >>= 1)
        contrib += __shfl_down(contrib, off, 32);
    if (lane == 0) wsum[wave] = contrib;
    __syncthreads();
    if (tid == 0) {
        float s = 0.0f;
#pragma unroll
        for (int w = 0; w < WAVES; ++w) s += wsum[w];
        partial[blockIdx.x] = s;   // fixed-order -> deterministic
    }
}

// ---------------------------------------------------------------------------
__global__ void finalize_loss_kernel(const float* __restrict__ partial,
                                     float* __restrict__ loss_out)
{
    float s = 0.0f;
    for (int i = 0; i < 512; ++i) s += partial[i];
    // vq_loss = (1 + COMMITMENT_COST) * sum / (BATCH * LAT)
    *loss_out = 1.1f * s * (1.0f / 6553600.0f);
}

// ---------------------------------------------------------------------------
// x_recon[row,:] = table[idx[row],:]   (pure bandwidth: 256 MB NT stores)
// ---------------------------------------------------------------------------
__global__ __launch_bounds__(256) void scatter_recon_kernel(
    const int* __restrict__ idx, const float* __restrict__ table,
    float* __restrict__ out)
{
    const int row = blockIdx.x;
    const int e = idx[row];
    const v4f* src = (const v4f*)(table + (size_t)e * IN_DIM);
    v4f* dst = (v4f*)(out + (size_t)row * IN_DIM);
    v4f val = src[threadIdx.x];
    __builtin_nontemporal_store(val, &dst[threadIdx.x]);
}

// ---------------------------------------------------------------------------
extern "C" void kernel_launch(void* const* d_in, const int* in_sizes, int n_in,
                              void* d_out, int out_size, void* d_ws, size_t ws_size,
                              hipStream_t stream)
{
    const float* x        = (const float*)d_in[0];
    const float* W_enc    = (const float*)d_in[1];
    const float* b_enc    = (const float*)d_in[2];
    const float* codebook = (const float*)d_in[3];
    const float* W_dec    = (const float*)d_in[4];
    const float* b_dec    = (const float*)d_in[5];

    float* outf  = (float*)d_out;
    float* recon = outf;                                     // 65536*1024
    float* loss  = outf + (size_t)BATCH * IN_DIM;            // 1
    int*   idxo  = (int*)(outf + (size_t)BATCH * IN_DIM + 1);// 65536

    float* ws      = (float*)d_ws;
    float* partial = ws;          // 512 floats (fully overwritten each call)
    float* table   = ws + 1024;   // 21*1024 floats

    recon_table_kernel<<<NE * 4, 256, 0, stream>>>(codebook, W_dec, b_dec, table);
    encoder_vq_kernel<<<BATCH / 128, 128, 0, stream>>>(x, W_enc, b_enc, codebook,
                                                       idxo, partial);
    finalize_loss_kernel<<<1, 1, 0, stream>>>(partial, loss);
    scatter_recon_kernel<<<BATCH, 256, 0, stream>>>(idxo, table, recon);
}
